// s2_GAT_37074157699770
// MI455X (gfx1250) — matrix operand order repaired and checked by reading, not compiled, not run on hardware
//
#include <hip/hip_runtime.h>
#include <hip/hip_bf16.h>

#define NND 50000
#define NED 800000
#define HID 128
#define EDIM 16
#define NPAIR 10000
#define NEG_SLOPE 0.2f
#define EDGE_SCALE (1.0f/20.0f)

typedef _Float16 v8h  __attribute__((ext_vector_type(8)));
typedef _Float16 v16h __attribute__((ext_vector_type(16)));
typedef float    v8f  __attribute__((ext_vector_type(8)));
typedef unsigned int u32x4 __attribute__((ext_vector_type(4)));
typedef int i32x4v __attribute__((ext_vector_type(4)));
typedef int i32x8v __attribute__((ext_vector_type(8)));

// ---------------- atomics / reductions ----------------
__device__ __forceinline__ void atomAddF(float* p, float v) {
  __hip_atomic_fetch_add(p, v, __ATOMIC_RELAXED, __HIP_MEMORY_SCOPE_AGENT);
}
__device__ __forceinline__ void atomMaxU(unsigned* p, unsigned v) {
  __hip_atomic_fetch_max(p, v, __ATOMIC_RELAXED, __HIP_MEMORY_SCOPE_AGENT);
}
// order-preserving float<->uint encode for atomic max
__device__ __forceinline__ unsigned fenc(float x) {
  unsigned u = __float_as_uint(x);
  return (u & 0x80000000u) ? ~u : (u | 0x80000000u);
}
__device__ __forceinline__ float fdec(unsigned u) {
  return (u & 0x80000000u) ? __uint_as_float(u & 0x7FFFFFFFu) : __uint_as_float(~u);
}
__device__ __forceinline__ float wredSum(float v) {
#pragma unroll
  for (int o = 16; o > 0; o >>= 1) v += __shfl_xor(v, o, 32);
  return v;
}
__device__ __forceinline__ float wredMax(float v) {
#pragma unroll
  for (int o = 16; o > 0; o >>= 1) v = fmaxf(v, __shfl_xor(v, o, 32));
  return v;
}
__device__ __forceinline__ float lrelu(float v) { return v > 0.f ? v : NEG_SLOPE * v; }

// ---------------- Tensor Data Mover: 2D tile (rows x elems, 2B elems) -> LDS ----------------
// D# per CDNA5 ISA: group0 = {flags, lds_addr, global_addr, type=2},
// group1 = {data_size=1(2B), tensor_dim0/1, tile_dim0/1, tensor_dim0_stride}. 2D => groups 2/3 zero.
__device__ __forceinline__ void tdm_load_2d_to_lds(const void* gptr, unsigned lds_off,
                                                   int elems_per_row, int rows) {
  unsigned long long g = (unsigned long long)gptr;
  u32x4 g0;
  g0.x = 1u;                                                 // count=1, user descriptor
  g0.y = lds_off;                                            // LDS byte address
  g0.z = (unsigned)(g & 0xFFFFFFFFull);                      // global_addr[31:0]
  g0.w = (unsigned)((g >> 32) & 0x01FFFFFFull) | 0x80000000u; // global_addr[56:32] | type=2
  i32x8v g1;
  unsigned d0 = (unsigned)elems_per_row;                     // tensor_dim0 == tile_dim0 == stride
  unsigned d1 = (unsigned)rows;                              // tensor_dim1 == tile_dim1
  g1[0] = 0x00010000;                                        // wg_mask=0, data_size=1 (2 bytes)
  g1[1] = (int)((d0 & 0xFFFFu) << 16);                       // tensor_dim0[15:0] @bits63:48
  g1[2] = (int)(((d0 >> 16) & 0xFFFFu) | ((d1 & 0xFFFFu) << 16)); // dim0[31:16] | dim1[15:0]
  g1[3] = (int)(((d1 >> 16) & 0xFFFFu) | ((d0 & 0xFFFFu) << 16)); // dim1[31:16] | tile_dim0
  g1[4] = (int)(d1 & 0xFFFFu);                               // tile_dim1 | tile_dim2=0
  g1[5] = (int)d0;                                           // tensor_dim0_stride[31:0]
  g1[6] = 0;                                                 // stride hi | dim1_stride lo
  g1[7] = 0;
  i32x4v z = {0, 0, 0, 0};
#if defined(__clang_major__) && __clang_major__ >= 23
  i32x8v z8 = {0, 0, 0, 0, 0, 0, 0, 0};
  __builtin_amdgcn_tensor_load_to_lds(g0, g1, z, z, z8, 0);
#else
  __builtin_amdgcn_tensor_load_to_lds(g0, g1, z, z, 0);
#endif
}

// ---------------- cast ----------------
__global__ __launch_bounds__(256) void cast_f32_f16(const float* __restrict__ in,
                                                    _Float16* __restrict__ out, int n) {
  for (int i = blockIdx.x * blockDim.x + threadIdx.x; i < n; i += gridDim.x * blockDim.x)
    out[i] = (_Float16)in[i];
}

// ---------------- WMMA GEMM: Y[M,Nout] = X[M,K] * W[Nout,K]^T + bias ----------------
// A layout: lane(l): h=l>>4, row=l&15; elems 0..7 -> K=8h+e, 8..15 -> K=16+8h+(e-8)
// B layout: lane(l): col=l&15;         elems 0..7 -> K=8h+e, 8..15 -> K=16+8h+(e-8)
// D layout: lane(l): N=l&15, M = vgpr + 8h
template <int NK, int NT>
__global__ __launch_bounds__(256) void gemm_wmma_kernel(
    const _Float16* __restrict__ X, const _Float16* __restrict__ W,
    const float* __restrict__ bias, float* __restrict__ Y, int M, int relu) {
  constexpr int K = NK * 32;
  constexpr int Nout = NT * 16;
  extern __shared__ _Float16 shW[];  // dynamic LDS => offset 0, matches TDM descriptor

  // Stage W[Nout x K] into LDS via the Tensor Data Mover: wave 0 issues the DMA.
  if (threadIdx.x < 32) {
    tdm_load_2d_to_lds(W, 0u, K, Nout);
    __builtin_amdgcn_s_wait_tensorcnt(0);
  }
  __syncthreads();

  const int lane = threadIdx.x & 31;
  const int wave = threadIdx.x >> 5;
  const int h = lane >> 4, rc = lane & 15;
  const int m0 = blockIdx.x * 128 + wave * 16;
  int mrow = m0 + rc; if (mrow >= M) mrow = M - 1;
  const _Float16* xrow = X + (size_t)mrow * K;

  v16h a[NK];
#pragma unroll
  for (int kk = 0; kk < NK; ++kk) {
    const _Float16* p = xrow + kk * 32;
    v8h lo = *(const v8h*)(p + 8 * h);
    v8h hi = *(const v8h*)(p + 16 + 8 * h);
#pragma unroll
    for (int i = 0; i < 8; ++i) { a[kk][i] = lo[i]; a[kk][8 + i] = hi[i]; }
  }
  const bool full = (m0 + 16 <= M);
#pragma unroll
  for (int t = 0; t < NT; ++t) {
    const int n0 = t << 4;
    float bv = bias[n0 + rc];
    v8f acc;
#pragma unroll
    for (int i = 0; i < 8; ++i) acc[i] = bv;
    const _Float16* wrow = shW + (n0 + rc) * K;
#pragma unroll
    for (int kk = 0; kk < NK; ++kk) {
      const _Float16* p = wrow + kk * 32 + 8 * h;
      v8h lo = *(const v8h*)p;
      v8h hi = *(const v8h*)(p + 16);
      v16h b;
#pragma unroll
      for (int i = 0; i < 8; ++i) { b[i] = lo[i]; b[8 + i] = hi[i]; }
      acc = __builtin_amdgcn_wmma_f32_16x16x32_f16(false, a[kk], false, b,
                                                   (short)0, acc, false, false);
    }
    if (full) {
#pragma unroll
      for (int v = 0; v < 8; ++v) {
        float o = acc[v];
        o = (relu && o < 0.f) ? 0.f : o;
        Y[(size_t)(m0 + v + 8 * h) * Nout + n0 + rc] = o;
      }
    } else {
#pragma unroll
      for (int v = 0; v < 8; ++v) {
        int m = m0 + v + 8 * h;
        if (m < M) {
          float o = acc[v];
          o = (relu && o < 0.f) ? 0.f : o;
          Y[(size_t)m * Nout + n0 + rc] = o;
        }
      }
    }
  }
}

// ---------------- per-node attention logits (3 heads), leaky-relu applied ----------------
__global__ __launch_bounds__(256) void node_scores_kernel(
    const float* __restrict__ Y,
    const float* __restrict__ Wa1, const float* __restrict__ ba1,
    const float* __restrict__ Wa2, const float* __restrict__ ba2,
    const float* __restrict__ Wa3, const float* __restrict__ ba3,
    float* __restrict__ lr, int N) {
  const int lane = threadIdx.x & 31;
  const int wid = blockIdx.x * (blockDim.x >> 5) + (threadIdx.x >> 5);
  const int nw = gridDim.x * (blockDim.x >> 5);
  float4 w1 = ((const float4*)Wa1)[lane];
  float4 w2 = ((const float4*)Wa2)[lane];
  float4 w3 = ((const float4*)Wa3)[lane];
  float b1 = ba1[0], b2 = ba2[0], b3 = ba3[0];
  for (int n = wid; n < N; n += nw) {
    float4 y = ((const float4*)(Y + (size_t)n * HID))[lane];
    float d1 = y.x * w1.x + y.y * w1.y + y.z * w1.z + y.w * w1.w;
    float d2 = y.x * w2.x + y.y * w2.y + y.z * w2.z + y.w * w2.w;
    float d3 = y.x * w3.x + y.y * w3.y + y.z * w3.z + y.w * w3.w;
    d1 = wredSum(d1); d2 = wredSum(d2); d3 = wredSum(d3);
    if (lane == 0) {
      lr[n]         = lrelu(d1 + b1);
      lr[N + n]     = lrelu(d2 + b2);
      lr[2 * N + n] = lrelu(d3 + b3);
    }
  }
}

// RED: [0..2]=encoded max(u32), [3..5]=Z sums, [6]=beta_loop const
__global__ void init_red_kernel(const float* __restrict__ W_edge,
                                const float* __restrict__ b_edge,
                                float* __restrict__ red) {
  if (threadIdx.x == 0 && blockIdx.x == 0) {
    unsigned* ru = (unsigned*)red;
    ru[0] = 0u; ru[1] = 0u; ru[2] = 0u;
    red[3] = 0.f; red[4] = 0.f; red[5] = 0.f;
    float s = 0.f;
#pragma unroll
    for (int j = 0; j < EDIM; ++j) s += W_edge[j];
    red[6] = EDGE_SCALE * (s + b_edge[0]);
  }
}

// self-loops cover every node -> edge-max == node-max
__global__ __launch_bounds__(256) void max_nodes_kernel(const float* __restrict__ lr,
                                                        float* __restrict__ red, int N) {
  float m0 = -3.3e38f, m1 = -3.3e38f, m2 = -3.3e38f;
  for (int n = blockIdx.x * blockDim.x + threadIdx.x; n < N; n += gridDim.x * blockDim.x) {
    m0 = fmaxf(m0, lr[n]); m1 = fmaxf(m1, lr[N + n]); m2 = fmaxf(m2, lr[2 * N + n]);
  }
  m0 = wredMax(m0); m1 = wredMax(m1); m2 = wredMax(m2);
  if ((threadIdx.x & 31) == 0) {
    unsigned* ru = (unsigned*)red;
    atomMaxU(ru + 0, fenc(m0)); atomMaxU(ru + 1, fenc(m1)); atomMaxU(ru + 2, fenc(m2));
  }
}

__global__ __launch_bounds__(256) void sumexp_nodes_kernel(const float* __restrict__ lr,
                                                           float* __restrict__ red, int N) {
  const unsigned* ru = (const unsigned*)red;
  float m0 = fdec(ru[0]), m1 = fdec(ru[1]), m2 = fdec(ru[2]);
  float s0 = 0.f, s1 = 0.f, s2 = 0.f;
  for (int n = blockIdx.x * blockDim.x + threadIdx.x; n < N; n += gridDim.x * blockDim.x) {
    s0 += __expf(lr[n] - m0); s1 += __expf(lr[N + n] - m1); s2 += __expf(lr[2 * N + n] - m2);
  }
  s0 = wredSum(s0); s1 = wredSum(s1); s2 = wredSum(s2);
  if ((threadIdx.x & 31) == 0) {
    atomAddF(red + 3, s0); atomAddF(red + 4, s1); atomAddF(red + 5, s2);
  }
}

__global__ __launch_bounds__(256) void sumexp_edges_kernel(const float* __restrict__ lr,
                                                           const int* __restrict__ src,
                                                           float* __restrict__ red, int N, int E) {
  const unsigned* ru = (const unsigned*)red;
  float m0 = fdec(ru[0]), m1 = fdec(ru[1]), m2 = fdec(ru[2]);
  float s0 = 0.f, s1 = 0.f, s2 = 0.f;
  for (int e = blockIdx.x * blockDim.x + threadIdx.x; e < E; e += gridDim.x * blockDim.x) {
    int s = src[e];
    s0 += __expf(lr[s] - m0); s1 += __expf(lr[N + s] - m1); s2 += __expf(lr[2 * N + s] - m2);
  }
  s0 = wredSum(s0); s1 = wredSum(s1); s2 = wredSum(s2);
  if ((threadIdx.x & 31) == 0) {
    atomAddF(red + 3, s0); atomAddF(red + 4, s1); atomAddF(red + 5, s2);
  }
}

__global__ __launch_bounds__(256) void edge_coef_kernel(
    const float* __restrict__ lr, const int* __restrict__ src,
    const float* __restrict__ eattr, const float* __restrict__ W_edge,
    const float* __restrict__ b_edge, const float* __restrict__ red,
    float* __restrict__ alpha, float* __restrict__ beta, int N, int E) {
  const unsigned* ru = (const unsigned*)red;
  float m0 = fdec(ru[0]), m1 = fdec(ru[1]), m2 = fdec(ru[2]);
  float rz0 = 1.f / red[3], rz1 = 1.f / red[4], rz2 = 1.f / red[5];
  float be = b_edge[0];
  float we[EDIM];
#pragma unroll
  for (int j = 0; j < EDIM; ++j) we[j] = W_edge[j];
  for (int e = blockIdx.x * blockDim.x + threadIdx.x; e < E; e += gridDim.x * blockDim.x) {
    int s = src[e];
    float a = __expf(lr[s] - m0) * rz0 + __expf(lr[N + s] - m1) * rz1 +
              __expf(lr[2 * N + s] - m2) * rz2;
    alpha[e] = a * (1.f / 3.f);
    const float* ea = eattr + (size_t)e * EDIM;
    float d = 0.f;
#pragma unroll
    for (int j = 0; j < EDIM; ++j) d += ea[j] * we[j];
    beta[e] = EDGE_SCALE * (d + be);
  }
}

// one wave per edge: S[dst] += alpha*y[src] + beta  (hardware f32 atomics, L2 resident)
__global__ __launch_bounds__(256) void scatter_kernel(
    const float* __restrict__ Y, const int* __restrict__ src, const int* __restrict__ dst,
    const float* __restrict__ alpha, const float* __restrict__ beta,
    float* __restrict__ S, int E) {
  const int lane = threadIdx.x & 31;
  const int wid = blockIdx.x * (blockDim.x >> 5) + (threadIdx.x >> 5);
  const int nw = gridDim.x * (blockDim.x >> 5);
  for (int e = wid; e < E; e += nw) {
    if (e + nw < E) {  // prefetch next gather row (global_prefetch_b8)
      int sn = src[e + nw];
      __builtin_prefetch(Y + (size_t)sn * HID + lane * 4, 0, 0);
    }
    int s = src[e], d = dst[e];
    float a = alpha[e], b = beta[e];
    float4 y = ((const float4*)(Y + (size_t)s * HID))[lane];
    float* o = S + (size_t)d * HID + lane * 4;
    atomAddF(o + 0, fmaf(a, y.x, b));
    atomAddF(o + 1, fmaf(a, y.y, b));
    atomAddF(o + 2, fmaf(a, y.z, b));
    atomAddF(o + 3, fmaf(a, y.w, b));
  }
}

// out = S + alpha_loop*Y + beta_loop + bias  (+relu). Write f32 or f16.
__global__ __launch_bounds__(256) void finalize_kernel(
    const float* __restrict__ S, const float* __restrict__ Y,
    const float* __restrict__ lr, const float* __restrict__ red,
    const float* __restrict__ bias, float* __restrict__ outF,
    _Float16* __restrict__ outH, int N, int relu) {
  const int lane = threadIdx.x & 31;
  const int wid = blockIdx.x * (blockDim.x >> 5) + (threadIdx.x >> 5);
  const int nw = gridDim.x * (blockDim.x >> 5);
  const unsigned* ru = (const unsigned*)red;
  float m0 = fdec(ru[0]), m1 = fdec(ru[1]), m2 = fdec(ru[2]);
  float rz0 = 1.f / red[3], rz1 = 1.f / red[4], rz2 = 1.f / red[5];
  float bl = red[6];
  float4 bv = ((const float4*)bias)[lane];
  for (int n = wid; n < N; n += nw) {
    float al = (__expf(lr[n] - m0) * rz0 + __expf(lr[N + n] - m1) * rz1 +
                __expf(lr[2 * N + n] - m2) * rz2) * (1.f / 3.f);
    float4 s = ((const float4*)(S + (size_t)n * HID))[lane];
    float4 y = ((const float4*)(Y + (size_t)n * HID))[lane];
    float o0 = s.x + al * y.x + bl + bv.x;
    float o1 = s.y + al * y.y + bl + bv.y;
    float o2 = s.z + al * y.z + bl + bv.z;
    float o3 = s.w + al * y.w + bl + bv.w;
    if (relu) { o0 = fmaxf(o0, 0.f); o1 = fmaxf(o1, 0.f); o2 = fmaxf(o2, 0.f); o3 = fmaxf(o3, 0.f); }
    if (outH) {
      _Float16* o = outH + (size_t)n * HID + lane * 4;
      o[0] = (_Float16)o0; o[1] = (_Float16)o1; o[2] = (_Float16)o2; o[3] = (_Float16)o3;
    } else {
      float* o = outF + (size_t)n * HID + lane * 4;
      o[0] = o0; o[1] = o1; o[2] = o2; o[3] = o3;
    }
  }
}

// merge: XM[p] = concat(HL[labels[p,0]], HR[labels[p,1]]) cast f16
__global__ __launch_bounds__(256) void merge_kernel(
    const float* __restrict__ HL, const float* __restrict__ HR,
    const int* __restrict__ labels, _Float16* __restrict__ XM, int P) {
  const int lane = threadIdx.x & 31;
  const int wid = blockIdx.x * (blockDim.x >> 5) + (threadIdx.x >> 5);
  const int nw = gridDim.x * (blockDim.x >> 5);
  for (int p = wid; p < P; p += nw) {
    int a = labels[p * 3 + 0], b = labels[p * 3 + 1];
    float4 l = ((const float4*)(HL + (size_t)a * HID))[lane];
    float4 r = ((const float4*)(HR + (size_t)b * HID))[lane];
    _Float16* o = XM + (size_t)p * 256 + lane * 4;
    o[0] = (_Float16)l.x; o[1] = (_Float16)l.y; o[2] = (_Float16)l.z; o[3] = (_Float16)l.w;
    o[128] = (_Float16)r.x; o[129] = (_Float16)r.y; o[130] = (_Float16)r.z; o[131] = (_Float16)r.w;
  }
}

// fc2: out[p, 0..1] = H[p,:] . fc2W[o,:] + b[o]
__global__ __launch_bounds__(256) void fc2_kernel(
    const float* __restrict__ H, const float* __restrict__ W,
    const float* __restrict__ b, float* __restrict__ out, int P) {
  const int lane = threadIdx.x & 31;
  const int wid = blockIdx.x * (blockDim.x >> 5) + (threadIdx.x >> 5);
  const int nw = gridDim.x * (blockDim.x >> 5);
  float4 w0 = ((const float4*)W)[lane];
  float4 w1 = ((const float4*)(W + HID))[lane];
  float b0 = b[0], b1 = b[1];
  for (int p = wid; p < P; p += nw) {
    float4 h = ((const float4*)(H + (size_t)p * HID))[lane];
    float d0 = h.x * w0.x + h.y * w0.y + h.z * w0.z + h.w * w0.w;
    float d1 = h.x * w1.x + h.y * w1.y + h.z * w1.z + h.w * w1.w;
    d0 = wredSum(d0); d1 = wredSum(d1);
    if (lane == 0) { out[p * 2 + 0] = d0 + b0; out[p * 2 + 1] = d1 + b1; }
  }
}

// ---------------- host ----------------
struct ConvP {
  const float *W_edge, *W_neigh, *W_self, *Wa1, *Wa2, *Wa3;
  const float *b_edge, *b_neigh, *b_self, *ba1, *ba2, *ba3, *bias;
};

extern "C" void kernel_launch(void* const* d_in, const int* in_sizes, int n_in,
                              void* d_out, int out_size, void* d_ws, size_t ws_size,
                              hipStream_t stream) {
  const float* x_l    = (const float*)d_in[0];
  const int*   ei_l   = (const int*)d_in[1];
  const float* ea_l   = (const float*)d_in[2];
  const float* x_r    = (const float*)d_in[3];
  const int*   ei_r   = (const int*)d_in[4];
  const float* ea_r   = (const float*)d_in[5];
  const int*   labels = (const int*)d_in[6];

  // params: jax tree-flatten order (dict keys sorted):
  // convs: conv1_l, conv1_r, conv2_l, conv2_r; per conv (sorted keys):
  // W_edge, W_neigh, W_self, Wa1, Wa2, Wa3, b_edge, b_neigh, b_self, ba1, ba2, ba3, bias
  const float* leaf[56];
  if (n_in >= 7 + 56) {
    for (int i = 0; i < 56; ++i) leaf[i] = (const float*)d_in[7 + i];
  } else {
    static const int lsz[13] = {16, 16384, 16384, 128, 128, 128, 1, 128, 128, 1, 1, 1, 128};
    const float* p = (const float*)d_in[7];
    int li = 0;
    for (int c = 0; c < 4; ++c)
      for (int j = 0; j < 13; ++j) { leaf[li++] = p; p += lsz[j]; }
    leaf[52] = p; p += 128 * 256;   // fc1_W
    leaf[53] = p; p += 128;         // fc1_b
    leaf[54] = p; p += 2 * 128;     // fc2_W
    leaf[55] = p;                   // fc2_b
  }
  ConvP cv[4];
  for (int c = 0; c < 4; ++c) {
    const float* const* L = leaf + c * 13;
    cv[c] = {L[0], L[1], L[2], L[3], L[4], L[5], L[6], L[7], L[8], L[9], L[10], L[11], L[12]};
  }
  const float* fc1W = leaf[52]; const float* fc1b = leaf[53];
  const float* fc2W = leaf[54]; const float* fc2b = leaf[55];

  // workspace layout (256B aligned)
  char* ws = (char*)d_ws;
  size_t off = 0;
  auto alloc = [&](size_t bytes) -> char* {
    char* r = ws + off; off = (off + bytes + 255) & ~(size_t)255; return r;
  };
  float*    HL    = (float*)alloc((size_t)NND * HID * 4);
  float*    HR    = (float*)alloc((size_t)NND * HID * 4);
  float*    S     = (float*)alloc((size_t)NND * HID * 4);
  float*    Yb    = (float*)alloc((size_t)NND * HID * 4);
  _Float16* XH    = (_Float16*)alloc((size_t)NND * HID * 2);
  float*    ALPHA = (float*)alloc((size_t)NED * 4);
  float*    BETA  = (float*)alloc((size_t)NED * 4);
  float*    LR    = (float*)alloc((size_t)3 * NND * 4);
  _Float16* WH    = (_Float16*)alloc((size_t)65536 * 2);
  float*    RED   = (float*)alloc(256 * 4);
  if (off > ws_size) return;

  const int TB = 256;
  auto layer = [&](const float* xin, bool xin_in_XH, const int* src, const int* dst,
                   const float* eattr, const ConvP& P, float* outF, _Float16* outH, int relu) {
    if (!xin_in_XH)
      cast_f32_f16<<<512, TB, 0, stream>>>(xin, XH, NND * HID);
    cast_f32_f16<<<64, TB, 0, stream>>>(P.W_self, WH, HID * HID);
    cast_f32_f16<<<64, TB, 0, stream>>>(P.W_neigh, WH + 16384, HID * HID);
    int gb = (NND + 127) / 128;
    gemm_wmma_kernel<4, 8><<<gb, TB, 32768, stream>>>(XH, WH, P.b_self, S, NND, 0);
    gemm_wmma_kernel<4, 8><<<gb, TB, 32768, stream>>>(XH, WH + 16384, P.b_neigh, Yb, NND, 0);
    node_scores_kernel<<<1024, TB, 0, stream>>>(Yb, P.Wa1, P.ba1, P.Wa2, P.ba2, P.Wa3, P.ba3,
                                                LR, NND);
    init_red_kernel<<<1, 64, 0, stream>>>(P.W_edge, P.b_edge, RED);
    max_nodes_kernel<<<512, TB, 0, stream>>>(LR, RED, NND);
    sumexp_nodes_kernel<<<512, TB, 0, stream>>>(LR, RED, NND);
    sumexp_edges_kernel<<<1024, TB, 0, stream>>>(LR, src, RED, NND, NED);
    edge_coef_kernel<<<1024, TB, 0, stream>>>(LR, src, eattr, P.W_edge, P.b_edge, RED,
                                              ALPHA, BETA, NND, NED);
    scatter_kernel<<<4096, TB, 0, stream>>>(Yb, src, dst, ALPHA, BETA, S, NED);
    finalize_kernel<<<1024, TB, 0, stream>>>(S, Yb, LR, RED, P.bias, outF, outH, NND, relu);
  };

  // Left branch: conv1_l (cv[0]) -> relu -> conv2_l (cv[2])
  layer(x_l, false, ei_l, ei_l + NED, ea_l, cv[0], nullptr, XH, 1);
  layer(nullptr, true, ei_l, ei_l + NED, ea_l, cv[2], HL, nullptr, 0);
  // Right branch: conv1_r (cv[1]) -> relu -> conv2_r (cv[3])
  layer(x_r, false, ei_r, ei_r + NED, ea_r, cv[1], nullptr, XH, 1);
  layer(nullptr, true, ei_r, ei_r + NED, ea_r, cv[3], HR, nullptr, 0);

  // merge + MLP head
  merge_kernel<<<512, TB, 0, stream>>>(HL, HR, labels, XH, NPAIR);
  cast_f32_f16<<<128, TB, 0, stream>>>(fc1W, WH, HID * 2 * HID);
  gemm_wmma_kernel<8, 8><<<(NPAIR + 127) / 128, TB, 65536, stream>>>(XH, WH, fc1b, S, NPAIR, 1);
  fc2_kernel<<<512, TB, 0, stream>>>(S, fc2W, fc2b, (float*)d_out, NPAIR);
}
